// ExpertChoiceMoE_7267084665537
// MI455X (gfx1250) — matrix operand order repaired and checked by reading, not compiled, not run on hardware
//
#include <hip/hip_runtime.h>
#include <hip/hip_bf16.h>

// ---------------------------------------------------------------------------
// Expert-Choice MoE forward for gfx1250 (MI455X), bf16 WMMA compute path.
// Outputs (concatenated flat, f32): results [16384,1024], router_logits
// [16384,8], selected_tokens [8,4096] (indices written as float values).
// ---------------------------------------------------------------------------

#define N_TOK 16384
#define DIM   1024
#define DFF_  4096
#define NEXP  8
#define KSEL  4096
#define TM    128
#define TN    128
#define KS    32

typedef __attribute__((ext_vector_type(16))) __bf16 bf16x16;
typedef __attribute__((ext_vector_type(8)))  float  f32x8;

union FragA { bf16x16 v; unsigned short u[16]; };

__device__ __forceinline__ unsigned short f2bf(float f) {
    unsigned u = __float_as_uint(f);
    unsigned r = u + 0x7FFFu + ((u >> 16) & 1u);   // round-to-nearest-even
    return (unsigned short)(r >> 16);
}

// Branchless tanh-GELU: v_exp_f32 + v_rcp_f32, no EXEC divergence.
__device__ __forceinline__ float gelu_fast(float x) {
    float t  = 0.7978845608028654f * (x + 0.044715f * x * x * x);
    float a  = fabsf(t);
    float e  = __expf(-2.0f * a);
    float th = (1.0f - e) * __builtin_amdgcn_rcpf(1.0f + e);
    th = __builtin_copysignf(th, t);
    return 0.5f * x * (1.0f + th);
}

// Low 32 bits of the generic address of a __shared__ object = LDS byte offset
// (flat LDS aperture maps addr[31:0] to the LDS address, ISA 10.2).
__device__ __forceinline__ unsigned lds_off(const void* p) {
    return (unsigned)(size_t)p;
}

// Async global -> LDS copy, 8 bytes per lane (VGLOBAL encoding, GV mode).
__device__ __forceinline__ void async_copy_b64(unsigned lds_dst, const void* gsrc) {
    asm volatile("global_load_async_to_lds_b64 %0, %1, off"
                 :: "v"(lds_dst), "v"((unsigned long long)(size_t)gsrc)
                 : "memory");
}
__device__ __forceinline__ void wait_async0() {
    asm volatile("s_wait_asynccnt 0" ::: "memory");
}

// ---------------------------------------------------------------------------
// Zero-fill for the results region (harness poisons d_out with 0xAA).
// ---------------------------------------------------------------------------
__global__ void zero_kernel(float* __restrict__ p, size_t n) {
    size_t i = (size_t)blockIdx.x * blockDim.x + threadIdx.x;
    if (i < n) p[i] = 0.0f;
}

// ---------------------------------------------------------------------------
// Router: one wave32 per token. 8 dots of length D, softmax in f32.
// Emits logits (f32) and per-expert sort keys (prob_bits<<32)|~token.
// ---------------------------------------------------------------------------
__global__ __launch_bounds__(256)
void router_kernel(const float* __restrict__ x, const float* __restrict__ wr,
                   float* __restrict__ logits_out,
                   unsigned long long* __restrict__ keys) {
    int wave = (int)((blockIdx.x * blockDim.x + threadIdx.x) >> 5);
    int lane = threadIdx.x & 31;
    if (wave >= N_TOK) return;
    const float* xr = x + (size_t)wave * DIM;

    float acc[NEXP];
#pragma unroll
    for (int e = 0; e < NEXP; ++e) acc[e] = 0.0f;

    for (int t = lane; t < DIM; t += 32) {
        float xv = xr[t];
#pragma unroll
        for (int e = 0; e < NEXP; ++e) acc[e] += xv * wr[e * DIM + t];
    }
#pragma unroll
    for (int e = 0; e < NEXP; ++e)
#pragma unroll
        for (int off = 16; off > 0; off >>= 1)
            acc[e] += __shfl_xor(acc[e], off, 32);

    float m = acc[0];
#pragma unroll
    for (int e = 1; e < NEXP; ++e) m = fmaxf(m, acc[e]);
    float p[NEXP], s = 0.0f;
#pragma unroll
    for (int e = 0; e < NEXP; ++e) { p[e] = expf(acc[e] - m); s += p[e]; }
    float inv = 1.0f / s;

    if (lane < NEXP) {
        logits_out[(size_t)wave * NEXP + lane] = acc[lane];
        float prob = p[lane] * inv;
        unsigned long long key =
            ((unsigned long long)__float_as_uint(prob) << 32) |
            (unsigned long long)(0xFFFFFFFFu - (unsigned)wave);
        keys[(size_t)lane * N_TOK + wave] = key;
    }
}

// ---------------------------------------------------------------------------
// Bitonic sort (descending) over 8 independent 16384-key segments.
// Global stage kernel: used only for compare distance j > 2048.
// ---------------------------------------------------------------------------
__global__ void bitonic_stage(unsigned long long* __restrict__ keys,
                              int n, int j, int k) {
    int i  = blockIdx.x * blockDim.x + threadIdx.x;
    if (i >= NEXP * n) return;
    int il = i & (n - 1);
    int e  = i / n;
    int pr = il ^ j;
    if (pr <= il) return;
    unsigned long long* seg = keys + (size_t)e * n;
    unsigned long long a = seg[il], b = seg[pr];
    bool up = ((il & k) == 0);
    bool sw = up ? (a < b) : (a > b);          // descending overall
    if (sw) { seg[il] = b; seg[pr] = a; }
}

// LDS stage kernel: runs all stages j = jstart..1 of phase k on a 4096-key
// chunk held in LDS (chunks never straddle expert segments).
__global__ __launch_bounds__(1024)
void bitonic_local(unsigned long long* __restrict__ keys, int k, int jstart) {
    __shared__ unsigned long long buf[4096];
    int base = blockIdx.x * 4096;
    for (int t = threadIdx.x; t < 4096; t += 1024) buf[t] = keys[base + t];
    __syncthreads();
    int il_base = base & (N_TOK - 1);          // offset within expert segment
    for (int j = jstart; j > 0; j >>= 1) {
        for (int t = threadIdx.x; t < 2048; t += 1024) {
            int i = ((t & ~(j - 1)) << 1) | (t & (j - 1));
            int p = i | j;
            unsigned long long a = buf[i], b = buf[p];
            bool up = (((il_base + i) & k) == 0);
            bool sw = up ? (a < b) : (a > b);
            if (sw) { buf[i] = b; buf[p] = a; }
        }
        __syncthreads();
    }
    for (int t = threadIdx.x; t < 4096; t += 1024) keys[base + t] = buf[t];
}

// ---------------------------------------------------------------------------
// Select top-KSEL per expert from sorted keys; emit gather indices, weights,
// and the selected_tokens output (as float in the f32 output buffer).
// ---------------------------------------------------------------------------
__global__ void select_kernel(const unsigned long long* __restrict__ keys,
                              int* __restrict__ sel_idx,
                              float* __restrict__ sel_w,
                              float* __restrict__ sel_out) {
    int i = blockIdx.x * blockDim.x + threadIdx.x;
    if (i >= NEXP * KSEL) return;
    int e = i / KSEL, jx = i - e * KSEL;
    unsigned long long key = keys[(size_t)e * N_TOK + jx];
    unsigned idx = 0xFFFFFFFFu - (unsigned)(key & 0xFFFFFFFFull);
    sel_idx[i] = (int)idx;
    sel_w[i]   = __uint_as_float((unsigned)(key >> 32));
    sel_out[i] = (float)idx;
}

// ---------------------------------------------------------------------------
// GEMM1: h = gelu(Xg @ W1 + b1), Xg = gathered tokens (f32 -> bf16).
// 128x128 tile / block, 8 waves, each wave 64x32 = 8 WMMA f32 accumulators.
// Software-pipelined: next K-tile staged in VGPRs during WMMA compute.
// ---------------------------------------------------------------------------
__global__ __launch_bounds__(256)
void gemm1_kernel(const float* __restrict__ x,
                  const float* __restrict__ W1,   // [DIM, DFF_] (one expert)
                  const float* __restrict__ b1,   // [DFF_]
                  const int* __restrict__ rows,   // [KSEL]
                  unsigned short* __restrict__ h) // [KSEL, DFF_] bf16 bits
{
    __shared__ unsigned short Abuf[TM][KS + 2];
    __shared__ unsigned short Bbuf[TN][KS + 2];

    int tid = threadIdx.x;
    int m0 = blockIdx.y * TM;
    int n0 = blockIdx.x * TN;

    int w = tid >> 5, lane = tid & 31;
    int wm = (w & 1) * 64;
    int wn = (w >> 1) * 32;
    int lane16 = lane & 15;
    int halfA  = (lane >> 4) * 8;
    int halfB  = (lane >> 4) * 16;

    int ar  = tid >> 3;            // A-tile row (this thread's first row)
    int ac  = (tid & 7) * 4;       // A-tile col
    int bkr = tid >> 5;            // B-tile k-row
    int bc  = lane * 4;            // B-tile col

    // Hoist gather row pointers into registers (loop-invariant).
    const float* xrow[4];
#pragma unroll
    for (int p = 0; p < 4; ++p)
        xrow[p] = x + (size_t)rows[m0 + ar + 32 * p] * DIM + ac;

    f32x8 acc[4][2];
#pragma unroll
    for (int a = 0; a < 4; ++a)
#pragma unroll
        for (int b = 0; b < 2; ++b)
#pragma unroll
            for (int v = 0; v < 8; ++v) acc[a][b][v] = 0.0f;

    float4 pa[4], pb[4];
#pragma unroll
    for (int p = 0; p < 4; ++p) {
        pa[p] = *(const float4*)(xrow[p]);
        pb[p] = *(const float4*)(W1 + (size_t)(bkr + 8 * p) * DFF_ + n0 + bc);
    }

    for (int k0 = 0; k0 < DIM; k0 += KS) {
        // commit staged tile to LDS
#pragma unroll
        for (int p = 0; p < 4; ++p) {
            int r = ar + 32 * p;
            Abuf[r][ac + 0] = f2bf(pa[p].x);
            Abuf[r][ac + 1] = f2bf(pa[p].y);
            Abuf[r][ac + 2] = f2bf(pa[p].z);
            Abuf[r][ac + 3] = f2bf(pa[p].w);
            int kr = bkr + 8 * p;
            Bbuf[bc + 0][kr] = f2bf(pb[p].x);
            Bbuf[bc + 1][kr] = f2bf(pb[p].y);
            Bbuf[bc + 2][kr] = f2bf(pb[p].z);
            Bbuf[bc + 3][kr] = f2bf(pb[p].w);
        }
        __syncthreads();

        // prefetch next K-tile into VGPRs (overlaps with WMMA below)
        int kn = k0 + KS;
        if (kn < DIM) {
#pragma unroll
            for (int p = 0; p < 4; ++p) {
                pa[p] = *(const float4*)(xrow[p] + kn);
                pb[p] = *(const float4*)(W1 + (size_t)(kn + bkr + 8 * p) * DFF_ + n0 + bc);
            }
            if (kn + KS < DIM)   // hint the weight stream 2 tiles ahead
                __builtin_prefetch(W1 + (size_t)(kn + KS + bkr) * DFF_ + n0 + bc, 0, 3);
        }

        FragA bfr[2];
#pragma unroll
        for (int nb = 0; nb < 2; ++nb) {
            int n = wn + nb * 16 + lane16;
#pragma unroll
            for (int ei = 0; ei < 16; ++ei) bfr[nb].u[ei] = Bbuf[n][((ei >> 3) << 4) + halfA + (ei & 7)];
        }
#pragma unroll
        for (int mb = 0; mb < 4; ++mb) {
            FragA afr;
            int m = wm + mb * 16 + lane16;
#pragma unroll
            for (int ei = 0; ei < 16; ++ei) {
                int kk = ((ei >> 3) << 4) + halfA + (ei & 7);
                afr.u[ei] = Abuf[m][kk];
            }
#pragma unroll
            for (int nb = 0; nb < 2; ++nb)
                acc[mb][nb] = __builtin_amdgcn_wmma_f32_16x16x32_bf16(
                    false, afr.v, false, bfr[nb].v, (short)0, acc[mb][nb],
                    false, false);
        }
        __syncthreads();
    }

    int rhalf = (lane >> 4) * 8;
#pragma unroll
    for (int mb = 0; mb < 4; ++mb) {
#pragma unroll
        for (int nb = 0; nb < 2; ++nb) {
            int n_abs = n0 + wn + nb * 16 + lane16;
            float bias = b1[n_abs];
#pragma unroll
            for (int v = 0; v < 8; ++v) {
                int m_abs = m0 + wm + mb * 16 + rhalf + v;
                float val = gelu_fast(acc[mb][nb][v] + bias);
                h[(size_t)m_abs * DFF_ + n_abs] = f2bf(val);
            }
        }
    }
}

// ---------------------------------------------------------------------------
// GEMM2: out_tile = (h @ W2 + b2) * weight, scatter-added into results.
// A operand (h, bf16) is streamed with GLOBAL_LOAD_ASYNC_TO_LDS_B64 into a
// ping-pong LDS buffer (ASYNCcnt-tracked); B is converted through VGPRs.
// ---------------------------------------------------------------------------
__global__ __launch_bounds__(256)
void gemm2_kernel(const unsigned short* __restrict__ h, // [KSEL, DFF_] bf16
                  const float* __restrict__ W2,         // [DFF_, DIM] (one expert)
                  const float* __restrict__ b2,         // [DIM]
                  const int* __restrict__ rows,         // [KSEL]
                  const float* __restrict__ wts,        // [KSEL]
                  float* __restrict__ results)          // [N_TOK, DIM]
{
    __shared__ unsigned short Abuf[2][TM][KS + 2];
    __shared__ unsigned short Bbuf[TN][KS + 2];
    __shared__ int   rowIdx[TM];
    __shared__ float rowW[TM];

    int tid = threadIdx.x;
    int m0 = blockIdx.y * TM;
    int n0 = blockIdx.x * TN;
    if (tid < TM) { rowIdx[tid] = rows[m0 + tid]; rowW[tid] = wts[m0 + tid]; }

    int w = tid >> 5, lane = tid & 31;
    int wm = (w & 1) * 64;
    int wn = (w >> 1) * 32;
    int lane16 = lane & 15;
    int halfA  = (lane >> 4) * 8;
    int halfB  = (lane >> 4) * 16;

    int ar  = tid >> 3;
    int ac  = (tid & 7) * 4;
    int bkr = tid >> 5;
    int bc  = lane * 4;

    // Per-thread source pointers for the A (h) stream and ping-pong LDS slots.
    const unsigned short* hrow[4];
    unsigned ldsA0[4], ldsA1[4];
#pragma unroll
    for (int p = 0; p < 4; ++p) {
        hrow[p]  = h + (size_t)(m0 + ar + 32 * p) * DFF_ + ac;
        ldsA0[p] = lds_off(&Abuf[0][ar + 32 * p][ac]);
        ldsA1[p] = lds_off(&Abuf[1][ar + 32 * p][ac]);
    }

    f32x8 acc[4][2];
#pragma unroll
    for (int a = 0; a < 4; ++a)
#pragma unroll
        for (int b = 0; b < 2; ++b)
#pragma unroll
            for (int v = 0; v < 8; ++v) acc[a][b][v] = 0.0f;

    // Prologue: async A tile 0 into Abuf[0]; B tile 0 into VGPRs.
#pragma unroll
    for (int p = 0; p < 4; ++p) async_copy_b64(ldsA0[p], hrow[p]);
    float4 pb[4];
#pragma unroll
    for (int p = 0; p < 4; ++p)
        pb[p] = *(const float4*)(W2 + (size_t)(bkr + 8 * p) * DIM + n0 + bc);
    wait_async0();
    __syncthreads();   // Abuf[0] + rowIdx/rowW visible block-wide

    int cur = 0;
    for (int k0 = 0; k0 < DFF_; k0 += KS) {
        // commit staged B regs to LDS (Bbuf readers finished at last barrier)
#pragma unroll
        for (int p = 0; p < 4; ++p) {
            int kr = bkr + 8 * p;
            Bbuf[bc + 0][kr] = f2bf(pb[p].x);
            Bbuf[bc + 1][kr] = f2bf(pb[p].y);
            Bbuf[bc + 2][kr] = f2bf(pb[p].z);
            Bbuf[bc + 3][kr] = f2bf(pb[p].w);
        }
        __syncthreads();

        int kn = k0 + KS;
        if (kn < DFF_) {
            // async A next tile into the other LDS plane (overlaps WMMA)
#pragma unroll
            for (int p = 0; p < 4; ++p)
                async_copy_b64(cur ? ldsA0[p] : ldsA1[p], hrow[p] + kn);
            // B next tile into VGPRs
#pragma unroll
            for (int p = 0; p < 4; ++p)
                pb[p] = *(const float4*)(W2 + (size_t)(kn + bkr + 8 * p) * DIM + n0 + bc);
            if (kn + KS < DFF_)
                __builtin_prefetch(W2 + (size_t)(kn + KS + bkr) * DIM + n0 + bc, 0, 3);
        }

        FragA bfr[2];
#pragma unroll
        for (int nb = 0; nb < 2; ++nb) {
            int n = wn + nb * 16 + lane16;
#pragma unroll
            for (int ei = 0; ei < 16; ++ei) bfr[nb].u[ei] = Bbuf[n][((ei >> 3) << 4) + halfA + (ei & 7)];
        }
#pragma unroll
        for (int mb = 0; mb < 4; ++mb) {
            FragA afr;
            int m = wm + mb * 16 + lane16;
#pragma unroll
            for (int ei = 0; ei < 16; ++ei) {
                int kk = ((ei >> 3) << 4) + halfA + (ei & 7);
                afr.u[ei] = Abuf[cur][m][kk];
            }
#pragma unroll
            for (int nb = 0; nb < 2; ++nb)
                acc[mb][nb] = __builtin_amdgcn_wmma_f32_16x16x32_bf16(
                    false, afr.v, false, bfr[nb].v, (short)0, acc[mb][nb],
                    false, false);
        }

        wait_async0();       // this wave's async stores into next plane done
        __syncthreads();     // all waves done -> next plane readable, Bbuf reusable
        cur ^= 1;
    }

    int rhalf = (lane >> 4) * 8;
#pragma unroll
    for (int mb = 0; mb < 4; ++mb) {
#pragma unroll
        for (int nb = 0; nb < 2; ++nb) {
            int n_abs = n0 + wn + nb * 16 + lane16;
            float bias = b2[n_abs];
#pragma unroll
            for (int v = 0; v < 8; ++v) {
                int m_loc = wm + mb * 16 + rhalf + v;
                float val = (acc[mb][nb][v] + bias) * rowW[m_loc];
                atomicAdd(results + (size_t)rowIdx[m_loc] * DIM + n_abs, val);
            }
        }
    }
}

// ---------------------------------------------------------------------------
// Host-side launch sequence (graph-capture safe, stream-serialized).
// ---------------------------------------------------------------------------
extern "C" void kernel_launch(void* const* d_in, const int* in_sizes, int n_in,
                              void* d_out, int out_size, void* d_ws, size_t ws_size,
                              hipStream_t stream) {
    const float* x  = (const float*)d_in[0];   // [8,2048,1024]
    const float* wr = (const float*)d_in[1];   // [8,1024]
    const float* W1 = (const float*)d_in[2];   // [8,1024,4096]
    const float* b1 = (const float*)d_in[3];   // [8,4096]
    const float* W2 = (const float*)d_in[4];   // [8,4096,1024]
    const float* b2 = (const float*)d_in[5];   // [8,1024]

    float* out_res    = (float*)d_out;                      // [16384,1024]
    float* out_logits = out_res + (size_t)N_TOK * DIM;      // [16384,8]
    float* out_sel    = out_logits + (size_t)N_TOK * NEXP;  // [8,4096]

    char* ws = (char*)d_ws;
    unsigned long long* keys = (unsigned long long*)ws; ws += (size_t)NEXP * N_TOK * 8;
    int*   sel_idx = (int*)ws;   ws += (size_t)NEXP * KSEL * 4;
    float* sel_w   = (float*)ws; ws += (size_t)NEXP * KSEL * 4;
    unsigned short* hbuf = (unsigned short*)ws;             // [KSEL, DFF_] bf16, reused per expert

    // 1) zero the results region (poisoned by harness)
    {
        size_t n = (size_t)N_TOK * DIM;
        zero_kernel<<<(unsigned)((n + 255) / 256), 256, 0, stream>>>(out_res, n);
    }
    // 2) router logits + softmax + sort keys
    router_kernel<<<(N_TOK * 32) / 256, 256, 0, stream>>>(x, wr, out_logits, keys);

    // 3) per-expert descending bitonic sort: global stages only for j>2048,
    //    then one LDS kernel finishes all j<=2048 stages of each phase.
    for (int k = 2; k <= N_TOK; k <<= 1) {
        for (int j = k >> 1; j > 2048; j >>= 1)
            bitonic_stage<<<(NEXP * N_TOK + 255) / 256, 256, 0, stream>>>(keys, N_TOK, j, k);
        int jstart = (k >> 1) > 2048 ? 2048 : (k >> 1);
        bitonic_local<<<(NEXP * N_TOK) / 4096, 1024, 0, stream>>>(keys, k, jstart);
    }

    // 4) extract indices / weights / selected_tokens output
    select_kernel<<<(NEXP * KSEL + 255) / 256, 256, 0, stream>>>(keys, sel_idx, sel_w, out_sel);

    // 5) expert FFNs, sequential over experts (hbuf reuse serialized by stream)
    for (int e = 0; e < NEXP; ++e) {
        gemm1_kernel<<<dim3(DFF_ / TN, KSEL / TM), 256, 0, stream>>>(
            x, W1 + (size_t)e * DIM * DFF_, b1 + (size_t)e * DFF_,
            sel_idx + (size_t)e * KSEL, hbuf);
        gemm2_kernel<<<dim3(DIM / TN, KSEL / TM), 256, 0, stream>>>(
            hbuf, W2 + (size_t)e * DFF_ * DIM, b2 + (size_t)e * DIM,
            sel_idx + (size_t)e * KSEL, sel_w + (size_t)e * KSEL, out_res);
    }
}
